// bfn4MolEGNN_73452530696326
// MI455X (gfx1250) — matrix operand order repaired and checked by reading, not compiled, not run on hardware
//
#include <hip/hip_runtime.h>
#include <hip/hip_bf16.h>

// ---------------------------------------------------------------------------
// Types for CDNA5 WMMA
// ---------------------------------------------------------------------------
typedef __attribute__((ext_vector_type(16))) __bf16        v16bf;
typedef __attribute__((ext_vector_type(8)))  float         v8f;
typedef __attribute__((ext_vector_type(4)))  unsigned int  u32x4;

#define H_DIM 256
#define TM    32      // edge/node rows per workgroup tile
#define LDA1  520     // LDS leading dim for K=512 tiles (pad to kill bank conflicts)
#define LDA2  264     // LDS leading dim for K=256 tiles

__device__ __forceinline__ unsigned short f2bf(float f) {
    unsigned u = __float_as_uint(f);
    unsigned r = u + 0x7FFFu + ((u >> 16) & 1u);   // round-to-nearest-even
    return (unsigned short)(r >> 16);
}
__device__ __forceinline__ float bf2f(unsigned short s) {
    return __uint_as_float(((unsigned)s) << 16);
}
__device__ __forceinline__ float silu(float x) {
    return x / (1.0f + __expf(-x));
}

struct Frag { union { v16bf v; u32x4 q[2]; }; };

// A fragment (16x32 bf16 tile, row-major in LDS).  Per ISA 7.12.2:
// lanes 0-15 hold K in {kb..kb+7, kb+16..kb+23}; lanes 16-31 hold {+8,+24}.
__device__ __forceinline__ Frag ld_a(const unsigned short* tile, int ld, int row,
                                     int kb, int lane) {
    int l8 = ((lane >> 4) & 1) << 3;
    const unsigned short* p = tile + row * ld + kb + l8;
    Frag f;
    f.q[0] = *(const u32x4*)(p);
    f.q[1] = *(const u32x4*)(p + 16);
    return f;
}
// B fragment from transposed weights WT[H_DIM][K] (row = output column n).
// Same K order as A: lanes 0-15 hold K {kb..kb+7, kb+16..kb+23}; lanes 16-31 hold {+8,+24}.
__device__ __forceinline__ Frag ld_b(const unsigned short* WT, int K, int n,
                                     int kb, int lane) {
    int l16 = ((lane >> 4) & 1) << 3;
    const unsigned short* p = WT + (size_t)n * K + kb + l16;
    Frag f;
    f.q[0] = *(const u32x4*)(p);
    f.q[1] = *(const u32x4*)(p + 16);
    return f;
}

__device__ __forceinline__ v8f wmma_bf16(Frag a, Frag b, v8f c) {
    return __builtin_amdgcn_wmma_f32_16x16x32_bf16(
        false, a.v, false, b.v, (short)0, c, false, false);
}

__device__ __forceinline__ void zero_acc(v8f acc[2][2]) {
    v8f z = {0.f, 0.f, 0.f, 0.f, 0.f, 0.f, 0.f, 0.f};
    acc[0][0] = z; acc[0][1] = z; acc[1][0] = z; acc[1][1] = z;
}

// 32x32 output tile per wave: 2 M-tiles x 2 N-tiles, K-loop over KTOT.
template <int KTOT, int LDA>
__device__ __forceinline__ void gemm_block(const unsigned short* Atile,
                                           const unsigned short* WT,
                                           int nb, int lane, v8f acc[2][2]) {
    const int l15 = lane & 15;
#pragma unroll 2
    for (int ks = 0; ks < KTOT / 32; ++ks) {
        int kb = ks * 32;
        // stream weights through L2; hint next k-step
        __builtin_prefetch((const void*)(WT + (size_t)(nb + l15) * KTOT + kb + 32), 0, 0);
        Frag b0 = ld_b(WT, KTOT, nb + l15,      kb, lane);
        Frag b1 = ld_b(WT, KTOT, nb + 16 + l15, kb, lane);
#pragma unroll
        for (int mt = 0; mt < 2; ++mt) {
            Frag a = ld_a(Atile, LDA, mt * 16 + l15, kb, lane);
            acc[mt][0] = wmma_bf16(a, b0, acc[mt][0]);
            acc[mt][1] = wmma_bf16(a, b1, acc[mt][1]);
        }
    }
}

// ---------------------------------------------------------------------------
// Weight prep: fp32 [K][256] -> bf16 transposed [256][K]
// ---------------------------------------------------------------------------
__global__ __launch_bounds__(256) void transpose_bf16_kernel(
    const float* __restrict__ src, unsigned short* __restrict__ dst, int K) {
    int idx = blockIdx.x * 256 + threadIdx.x;
    if (idx < K * H_DIM) {
        int n = idx / K;
        int k = idx - n * K;
        dst[(size_t)n * K + k] = f2bf(src[(size_t)k * H_DIM + n]);
    }
}

// ---------------------------------------------------------------------------
// Embedding: gaussian basis (16) + time -> h = basis17 @ W + b ; copy x
// ---------------------------------------------------------------------------
__global__ __launch_bounds__(256) void embed_kernel(
    const float* __restrict__ timev, const float* __restrict__ muq,
    const float* __restrict__ mupos, const float* __restrict__ Wi,
    const float* __restrict__ bi, float* __restrict__ h, float* __restrict__ x,
    int Ntot) {
    int i = blockIdx.x;
    if (i >= Ntot) return;
    __shared__ float sret[16];
    __shared__ float sden;
    int t = threadIdx.x;
    float q = muq[i];
    if (t < 16) {
        float c = -2.0f + (4.0f / 15.0f) * (float)t;  // linspace(-2,2,16)
        float o = (q - c) * 4.0f;                     // / WIDTH, WIDTH = 0.25
        sret[t] = __expf(-0.5f * o * o);
    }
    __syncthreads();
    if (t == 0) {
        float d = 0.f;
        for (int j = 0; j < 16; ++j) d += fabsf(sret[j]);
        sden = fmaxf(d, 1e-12f);
    }
    __syncthreads();
    float den = sden;
    float s = bi[t];
#pragma unroll
    for (int f = 0; f < 16; ++f) {
        float feat = sret[f] / den * 2.0f - 1.0f;
        s += feat * Wi[f * H_DIM + t];
    }
    s += timev[i] * Wi[16 * H_DIM + t];
    h[(size_t)i * H_DIM + t] = s;
    if (t < 3) x[i * 3 + t] = mupos[i * 3 + t];
}

// ---------------------------------------------------------------------------
// Edge degree count (on row index)
// ---------------------------------------------------------------------------
__global__ __launch_bounds__(256) void count_kernel(
    const int* __restrict__ eidx, float* __restrict__ cnt, int Etot) {
    int e = blockIdx.x * 256 + threadIdx.x;
    if (e < Etot) atomicAdd(&cnt[eidx[e]], 1.0f);
}

// ---------------------------------------------------------------------------
// Fused edge-MLP kernel: the hot loop.  32 edges per workgroup, 8 wave32s.
// ---------------------------------------------------------------------------
__global__ __launch_bounds__(256) void edge_kernel(
    const float* __restrict__ h, const float* __restrict__ x,
    const int* __restrict__ eidx, int Etot,
    const unsigned short* __restrict__ w1T, const float* __restrict__ w1last,
    const float* __restrict__ b1,
    const unsigned short* __restrict__ w2T, const float* __restrict__ b2,
    const unsigned short* __restrict__ cw1T, const float* __restrict__ cb1,
    const float* __restrict__ cw2,
    float* __restrict__ agg, float* __restrict__ xacc) {
    __shared__ __align__(16) unsigned short bufA[TM * LDA1];
    __shared__ __align__(16) unsigned short bufB[TM * LDA2];
    __shared__ int   srow[TM], scol[TM];
    __shared__ float sdiff[TM][3];
    __shared__ float srad[TM];
    __shared__ float sdot[TM][4];

    const int tid  = threadIdx.x;
    const int lane = tid & 31;
    const int wave = tid >> 5;
    const int nb   = wave * 32;      // wave's 32-column slice of the 256 outputs
    const int l15  = lane & 15;
    const int rhi  = (lane >> 4) << 3;
    const int e0   = blockIdx.x * TM;

    if (tid < TM) {
        int e = e0 + tid;
        int r = eidx[e];
        int c = eidx[Etot + e];
        srow[tid] = r;
        scol[tid] = c;
        float dx = x[r * 3 + 0] - x[c * 3 + 0];
        float dy = x[r * 3 + 1] - x[c * 3 + 1];
        float dz = x[r * 3 + 2] - x[c * 3 + 2];
        sdiff[tid][0] = dx; sdiff[tid][1] = dy; sdiff[tid][2] = dz;
        srad[tid] = dx * dx + dy * dy + dz * dz;
    }
    __syncthreads();

    // Gather A = [h[row] || h[col]] as bf16, 32x512
    for (int idx = tid; idx < TM * 512; idx += 256) {
        int r = idx >> 9;
        int c = idx & 511;
        int node = (c < H_DIM) ? srow[r] : scol[r];
        bufA[r * LDA1 + c] = f2bf(h[(size_t)node * H_DIM + (c & (H_DIM - 1))]);
    }
    __syncthreads();

    v8f acc[2][2];

    // ---- GEMM1: t1 = silu(e_in @ w1 + b1), radial handled as rank-1 term
    zero_acc(acc);
    gemm_block<512, LDA1>(bufA, w1T, nb, lane, acc);
#pragma unroll
    for (int mt = 0; mt < 2; ++mt)
#pragma unroll
        for (int nt = 0; nt < 2; ++nt) {
            int col = nb + nt * 16 + l15;
            float wl = w1last[col], bb = b1[col];
#pragma unroll
            for (int i = 0; i < 8; ++i) {
                int row = mt * 16 + i + rhi;
                float v = acc[mt][nt][i] + srad[row] * wl + bb;
                bufB[row * LDA2 + col] = f2bf(silu(v));
            }
        }
    __syncthreads();

    // ---- GEMM2: m = silu(t1 @ w2 + b2); scatter-add m into agg[row]
    zero_acc(acc);
    gemm_block<256, LDA2>(bufB, w2T, nb, lane, acc);
#pragma unroll
    for (int mt = 0; mt < 2; ++mt)
#pragma unroll
        for (int nt = 0; nt < 2; ++nt) {
            int col = nb + nt * 16 + l15;
            float bb = b2[col];
#pragma unroll
            for (int i = 0; i < 8; ++i) {
                int row = mt * 16 + i + rhi;
                float v = silu(acc[mt][nt][i] + bb);
                bufA[row * LDA2 + col] = f2bf(v);   // reuse bufA as m-tile
                atomicAdd(&agg[(size_t)srow[row] * H_DIM + col], v);
            }
        }
    __syncthreads();

    // ---- GEMM3: c1 = silu(m @ cw1 + cb1)
    zero_acc(acc);
    gemm_block<256, LDA2>(bufA, cw1T, nb, lane, acc);
#pragma unroll
    for (int mt = 0; mt < 2; ++mt)
#pragma unroll
        for (int nt = 0; nt < 2; ++nt) {
            int col = nb + nt * 16 + l15;
            float bb = cb1[col];
#pragma unroll
            for (int i = 0; i < 8; ++i) {
                int row = mt * 16 + i + rhi;
                bufB[row * LDA2 + col] = f2bf(silu(acc[mt][nt][i] + bb));
            }
        }
    __syncthreads();

    // ---- s = c1 @ cw2 (256 -> 1), trans = diff * s, scatter into xacc
    if (tid < TM * 4) {
        int r = tid >> 2, j = tid & 3;
        const unsigned short* crow = bufB + r * LDA2 + j * 64;
        const float* wv = cw2 + j * 64;
        float s = 0.f;
#pragma unroll 8
        for (int k = 0; k < 64; ++k) s += bf2f(crow[k]) * wv[k];
        sdot[r][j] = s;
    }
    __syncthreads();
    if (tid < TM) {
        float s = sdot[tid][0] + sdot[tid][1] + sdot[tid][2] + sdot[tid][3];
        int node = srow[tid];
        atomicAdd(&xacc[node * 3 + 0], sdiff[tid][0] * s);
        atomicAdd(&xacc[node * 3 + 1], sdiff[tid][1] * s);
        atomicAdd(&xacc[node * 3 + 2], sdiff[tid][2] * s);
    }
}

// ---------------------------------------------------------------------------
// x += xacc / max(cnt,1);  zero xacc for next layer
// ---------------------------------------------------------------------------
__global__ __launch_bounds__(256) void x_update_kernel(
    float* __restrict__ x, float* __restrict__ xacc,
    const float* __restrict__ cnt, int Ntot) {
    int i = blockIdx.x * 256 + threadIdx.x;
    if (i < Ntot * 3) {
        float c = fmaxf(cnt[i / 3], 1.0f);
        x[i] += xacc[i] / c;
        xacc[i] = 0.0f;
    }
}

// ---------------------------------------------------------------------------
// Fused node-MLP: h += silu([h||agg] @ nw1 + nb1) @ nw2 + nb2 ; zero agg
// ---------------------------------------------------------------------------
__global__ __launch_bounds__(256) void node_kernel(
    float* __restrict__ h, float* __restrict__ agg,
    const unsigned short* __restrict__ nw1T, const float* __restrict__ nb1,
    const unsigned short* __restrict__ nw2T, const float* __restrict__ nb2,
    int Ntot) {
    __shared__ __align__(16) unsigned short bufA[TM * LDA1];
    __shared__ __align__(16) unsigned short bufB[TM * LDA2];
    const int tid  = threadIdx.x;
    const int lane = tid & 31;
    const int wave = tid >> 5;
    const int nb   = wave * 32;
    const int l15  = lane & 15;
    const int rhi  = (lane >> 4) << 3;
    const int n0   = blockIdx.x * TM;

    for (int idx = tid; idx < TM * 512; idx += 256) {
        int r = idx >> 9;
        int c = idx & 511;
        int node = n0 + r;
        float v = 0.f;
        if (node < Ntot) {
            if (c < H_DIM) {
                v = h[(size_t)node * H_DIM + c];
            } else {
                size_t a = (size_t)node * H_DIM + (c - H_DIM);
                v = agg[a];
                agg[a] = 0.0f;   // ready for next layer
            }
        }
        bufA[r * LDA1 + c] = f2bf(v);
    }
    __syncthreads();

    v8f acc[2][2];
    zero_acc(acc);
    gemm_block<512, LDA1>(bufA, nw1T, nb, lane, acc);
#pragma unroll
    for (int mt = 0; mt < 2; ++mt)
#pragma unroll
        for (int nt = 0; nt < 2; ++nt) {
            int col = nb + nt * 16 + l15;
            float bb = nb1[col];
#pragma unroll
            for (int i = 0; i < 8; ++i) {
                int row = mt * 16 + i + rhi;
                bufB[row * LDA2 + col] = f2bf(silu(acc[mt][nt][i] + bb));
            }
        }
    __syncthreads();

    zero_acc(acc);
    gemm_block<256, LDA2>(bufB, nw2T, nb, lane, acc);
#pragma unroll
    for (int mt = 0; mt < 2; ++mt)
#pragma unroll
        for (int nt = 0; nt < 2; ++nt) {
            int col = nb + nt * 16 + l15;
            float bb = nb2[col];
#pragma unroll
            for (int i = 0; i < 8; ++i) {
                int row = mt * 16 + i + rhi;
                int node = n0 + row;
                if (node < Ntot)
                    h[(size_t)node * H_DIM + col] += acc[mt][nt][i] + bb;
            }
        }
}

// ---------------------------------------------------------------------------
// Segment sums for zero-center-of-mass
// ---------------------------------------------------------------------------
__global__ __launch_bounds__(256) void segsum_kernel(
    const float* __restrict__ x, const float* __restrict__ mupos,
    const int* __restrict__ segid, float* __restrict__ ssum,
    float* __restrict__ scnt, int Ntot) {
    int i = blockIdx.x * 256 + threadIdx.x;
    if (i < Ntot) {
        int s = segid[i];
        atomicAdd(&ssum[s * 3 + 0], x[i * 3 + 0] - mupos[i * 3 + 0]);
        atomicAdd(&ssum[s * 3 + 1], x[i * 3 + 1] - mupos[i * 3 + 1]);
        atomicAdd(&ssum[s * 3 + 2], x[i * 3 + 2] - mupos[i * 3 + 2]);
        atomicAdd(&scnt[s], 1.0f);
    }
}

// ---------------------------------------------------------------------------
// Final epilogue: out = [coord_pred (N*3) | k_hat (N)]
// ---------------------------------------------------------------------------
__global__ __launch_bounds__(256) void final_kernel(
    const float* __restrict__ h, const float* __restrict__ x,
    const float* __restrict__ mupos, const float* __restrict__ muq,
    const float* __restrict__ gcoord, const float* __restrict__ gcharge,
    const int* __restrict__ segid, const float* __restrict__ ssum,
    const float* __restrict__ scnt, const float* __restrict__ Wo,
    const float* __restrict__ bo, float* __restrict__ out, int Ntot) {
    int node = blockIdx.x * 8 + (threadIdx.x >> 5);
    int lane = threadIdx.x & 31;
    if (node >= Ntot) return;
    float p = 0.f;
    for (int k = lane; k < H_DIM; k += 32)
        p += h[(size_t)node * H_DIM + k] * Wo[k * 2];   // column 0 = h[:, -2]
#pragma unroll
    for (int off = 16; off > 0; off >>= 1) p += __shfl_xor(p, off, 32);
    if (lane == 0) {
        float mce = p + bo[0];
        mce = fminf(10.f, fmaxf(-10.f, mce));
        float g = gcharge[node];
        out[(size_t)Ntot * 3 + node] = muq[node] / g - sqrtf((1.f - g) / g) * mce;
        int s = segid[node];
        float c = fmaxf(scnt[s], 1.0f);
        float gg = gcoord[node];
        float sc = sqrtf((1.f - gg) / gg);
        for (int d = 0; d < 3; ++d) {
            float eps = (x[node * 3 + d] - mupos[node * 3 + d]) - ssum[s * 3 + d] / c;
            eps = fminf(10.f, fmaxf(-10.f, eps));
            out[node * 3 + d] = mupos[node * 3 + d] / gg - sc * eps;
        }
    }
}

// ---------------------------------------------------------------------------
// Host-side orchestration
// ---------------------------------------------------------------------------
extern "C" void kernel_launch(void* const* d_in, const int* in_sizes, int n_in,
                              void* d_out, int out_size, void* d_ws, size_t ws_size,
                              hipStream_t stream) {
    (void)in_sizes; (void)n_in; (void)out_size; (void)ws_size;
    const int N = 20000, E = 320000, Bseg = 1000, Lc = 4;

    const float* timev  = (const float*)d_in[0];
    const float* muq    = (const float*)d_in[1];
    const float* mupos  = (const float*)d_in[2];
    const float* gcoord = (const float*)d_in[3];
    const float* gchrg  = (const float*)d_in[4];
    const int*   eidx   = (const int*)d_in[5];
    const int*   segid  = (const int*)d_in[6];
    const float* emb_in_w = (const float*)d_in[7];
    const float* emb_in_b = (const float*)d_in[8];
    const float* edge_w1  = (const float*)d_in[9];
    const float* edge_b1  = (const float*)d_in[10];
    const float* edge_w2  = (const float*)d_in[11];
    const float* edge_b2  = (const float*)d_in[12];
    const float* node_w1  = (const float*)d_in[13];
    const float* node_b1  = (const float*)d_in[14];
    const float* node_w2  = (const float*)d_in[15];
    const float* node_b2  = (const float*)d_in[16];
    const float* coord_w1 = (const float*)d_in[17];
    const float* coord_b1 = (const float*)d_in[18];
    const float* coord_w2 = (const float*)d_in[19];
    const float* emb_out_w = (const float*)d_in[20];
    const float* emb_out_b = (const float*)d_in[21];

    char*  ws  = (char*)d_ws;
    size_t off = 0;
    auto alloc = [&](size_t bytes) -> void* {
        void* p = ws + off;
        off = (off + bytes + 255) & ~(size_t)255;
        return p;
    };
    float* x    = (float*)alloc(sizeof(float) * (size_t)N * 3);
    float* h    = (float*)alloc(sizeof(float) * (size_t)N * H_DIM);
    float* agg  = (float*)alloc(sizeof(float) * (size_t)N * H_DIM);
    float* xacc = (float*)alloc(sizeof(float) * (size_t)N * 3);
    float* cnt  = (float*)alloc(sizeof(float) * (size_t)N);
    float* ssum = (float*)alloc(sizeof(float) * (size_t)Bseg * 3);
    float* scnt = (float*)alloc(sizeof(float) * (size_t)Bseg);
    unsigned short* w1T  = (unsigned short*)alloc(sizeof(short) * (size_t)Lc * 256 * 512);
    unsigned short* w2T  = (unsigned short*)alloc(sizeof(short) * (size_t)Lc * 256 * 256);
    unsigned short* cw1T = (unsigned short*)alloc(sizeof(short) * (size_t)Lc * 256 * 256);
    unsigned short* nw1T = (unsigned short*)alloc(sizeof(short) * (size_t)Lc * 256 * 512);
    unsigned short* nw2T = (unsigned short*)alloc(sizeof(short) * (size_t)Lc * 256 * 256);

    hipMemsetAsync(agg,  0, sizeof(float) * (size_t)N * H_DIM, stream);
    hipMemsetAsync(xacc, 0, sizeof(float) * (size_t)N * 3, stream);
    hipMemsetAsync(cnt,  0, sizeof(float) * (size_t)N, stream);
    hipMemsetAsync(ssum, 0, sizeof(float) * (size_t)Bseg * 3, stream);
    hipMemsetAsync(scnt, 0, sizeof(float) * (size_t)Bseg, stream);

    // Weight prep: bf16 transposed copies (L2-resident during the main loop)
    for (int l = 0; l < Lc; ++l) {
        transpose_bf16_kernel<<<512, 256, 0, stream>>>(
            edge_w1 + (size_t)l * 513 * 256, w1T + (size_t)l * 256 * 512, 512);
        transpose_bf16_kernel<<<256, 256, 0, stream>>>(
            edge_w2 + (size_t)l * 256 * 256, w2T + (size_t)l * 256 * 256, 256);
        transpose_bf16_kernel<<<256, 256, 0, stream>>>(
            coord_w1 + (size_t)l * 256 * 256, cw1T + (size_t)l * 256 * 256, 256);
        transpose_bf16_kernel<<<512, 256, 0, stream>>>(
            node_w1 + (size_t)l * 512 * 256, nw1T + (size_t)l * 256 * 512, 512);
        transpose_bf16_kernel<<<256, 256, 0, stream>>>(
            node_w2 + (size_t)l * 256 * 256, nw2T + (size_t)l * 256 * 256, 256);
    }

    embed_kernel<<<N, 256, 0, stream>>>(timev, muq, mupos, emb_in_w, emb_in_b,
                                        h, x, N);
    count_kernel<<<(E + 255) / 256, 256, 0, stream>>>(eidx, cnt, E);

    for (int l = 0; l < Lc; ++l) {
        edge_kernel<<<E / TM, 256, 0, stream>>>(
            h, x, eidx, E,
            w1T + (size_t)l * 256 * 512,
            edge_w1 + (size_t)l * 513 * 256 + (size_t)512 * 256,  // radial row
            edge_b1 + l * 256,
            w2T + (size_t)l * 256 * 256, edge_b2 + l * 256,
            cw1T + (size_t)l * 256 * 256, coord_b1 + l * 256,
            coord_w2 + (size_t)l * 256,
            agg, xacc);
        x_update_kernel<<<(N * 3 + 255) / 256, 256, 0, stream>>>(x, xacc, cnt, N);
        node_kernel<<<(N + TM - 1) / TM, 256, 0, stream>>>(
            h, agg,
            nw1T + (size_t)l * 256 * 512, node_b1 + l * 256,
            nw2T + (size_t)l * 256 * 256, node_b2 + l * 256, N);
    }

    segsum_kernel<<<(N + 255) / 256, 256, 0, stream>>>(x, mupos, segid, ssum, scnt, N);
    final_kernel<<<(N + 7) / 8, 256, 0, stream>>>(
        h, x, mupos, muq, gcoord, gchrg, segid, ssum, scnt,
        emb_out_w, emb_out_b, (float*)d_out, N);
}
